// AttLayer_37220186587843
// MI455X (gfx1250) — matrix operand order repaired and checked by reading, not compiled, not run on hardware
//
#include <hip/hip_runtime.h>
#include <hip/hip_bf16.h>
#include <cstdint>
#include <cstddef>

// ---------------------------------------------------------------------------
// AttLayer for MI455X (gfx1250): f16 WMMA pipeline with flash-attention core.
//   B=8, C=512, L=2048, CP=256
// ---------------------------------------------------------------------------

#define DEVI __device__ __forceinline__

typedef _Float16 half16 __attribute__((ext_vector_type(16)));
typedef float    float8 __attribute__((ext_vector_type(8)));

constexpr int B_  = 8;
constexpr int C_  = 512;
constexpr int L_  = 2048;
constexpr int CP_ = 256;

union FragU { half16 h; uint4 u[2]; };

DEVI int lane_id() { return (int)(threadIdx.x & 31u); }

// A-operand fragment (16x32 f16). Source must be row-major [M][K] (K contiguous).
// ISA layout: lane&15 = row M; lanes 0-15 take K {0..7,16..23}, lanes 16-31 take
// K {8..15,24..31}  ->  two 16-byte loads per lane.
DEVI half16 load_a_frag(const _Float16* p, int ld, int row0, int k0) {
  int lane = lane_id();
  int r  = row0 + (lane & 15);
  int hi = lane >> 4;
  const _Float16* base = p + (size_t)r * ld + k0 + hi * 8;
  FragU f;
  f.u[0] = *(const uint4*)(base);
  f.u[1] = *(const uint4*)(base + 16);
  return f.h;
}

// B-operand fragment (32x16 f16). Source must be [N][K] (K contiguous).
// ISA layout: lane&15 = col N; lanes 0-15 take K {0..7,16..23}, lanes 16-31 take
// K {8..15,24..31}  ->  two 16-byte loads per lane.
DEVI half16 load_b_frag(const _Float16* p, int ld, int col0, int k0) {
  int lane = lane_id();
  int n   = col0 + (lane & 15);
  int khi = lane >> 4;
  const _Float16* base = p + (size_t)n * ld + k0 + khi * 8;
  FragU f;
  f.u[0] = *(const uint4*)(base);
  f.u[1] = *(const uint4*)(base + 16);
  return f.h;
}

DEVI float8 wmma_f16(half16 a, half16 b, float8 c) {
  return __builtin_amdgcn_wmma_f32_16x16x32_f16(
      /*neg_a=*/false, a, /*neg_b=*/false, b,
      /*c_mod=*/(short)0, c, /*reuse_a=*/false, /*reuse_b=*/false);
}

// ---------------------------------------------------------------------------
// Kernel 1: f32 -> f16 copy (weights)
// ---------------------------------------------------------------------------
__global__ void cvt_f32_f16_kernel(const float* __restrict__ src,
                                   _Float16* __restrict__ dst, int n) {
  int i = blockIdx.x * blockDim.x + threadIdx.x;
  if (i < n) dst[i] = (_Float16)src[i];
}

// ---------------------------------------------------------------------------
// Kernel 2: x1 (B,C,L) f32  ->  xt (B,L,C) f16  (LDS tile transpose 32x32)
// ---------------------------------------------------------------------------
__global__ __launch_bounds__(256) void transpose_x_kernel(
    const float* __restrict__ x1, _Float16* __restrict__ xt) {
  __shared__ _Float16 t[32][33];
  int b = blockIdx.z;
  int lbase = blockIdx.x * 32;
  int cbase = blockIdx.y * 32;
  int tx = threadIdx.x, ty = threadIdx.y;
#pragma unroll
  for (int i = 0; i < 4; ++i) {
    int c = cbase + ty + i * 8;
    t[ty + i * 8][tx] = (_Float16)x1[((size_t)b * C_ + c) * L_ + lbase + tx];
  }
  __syncthreads();
#pragma unroll
  for (int i = 0; i < 4; ++i) {
    int l = lbase + ty + i * 8;
    xt[((size_t)b * L_ + l) * C_ + cbase + tx] = t[tx][ty + i * 8];
  }
}

// ---------------------------------------------------------------------------
// Kernel 3: QKV projection GEMM.
//   q/k/v[o,l] = sum_c W[o,c] * x[c,l] + bias[o]
//   Per wave: 16(M=CP dim) x 64(N=L dim) strip, K=C=512 in steps of 32.
//   Q is pre-scaled by 1/sqrt(CP) and stored transposed (L,CP), K stored
//   transposed (L,CP), V stored (CP,L) — feeding the flash kernel's operand
//   layouts directly.
// ---------------------------------------------------------------------------
__global__ __launch_bounds__(256) void qkv_gemm_kernel(
    const _Float16* __restrict__ xt, const _Float16* __restrict__ wh,
    const float* __restrict__ bq, const float* __restrict__ bk,
    const float* __restrict__ bv,
    _Float16* __restrict__ qt, _Float16* __restrict__ kt,
    _Float16* __restrict__ vv) {
  int w = (int)((blockIdx.x * blockDim.x + threadIdx.x) >> 5);
  int nt = w & 31;  w >>= 5;        // L/64  = 32 tiles
  int mt = w & 15;  w >>= 4;        // CP/16 = 16 tiles
  int b  = w & 7;   w >>= 3;        // batch
  int mat = w;                      // 0=Q,1=K,2=V

  const _Float16* W = wh + (size_t)mat * CP_ * C_;
  const float* bias = (mat == 0) ? bq : (mat == 1) ? bk : bv;
  const _Float16* X = xt + (size_t)b * L_ * C_;
  int m0 = mt * 16, n0 = nt * 64;

  float8 acc[4] = {};
  for (int k0 = 0; k0 < C_; k0 += 32) {
    half16 a = load_a_frag(W, C_, m0, k0);
#pragma unroll
    for (int j = 0; j < 4; ++j)
      acc[j] = wmma_f16(a, load_b_frag(X, C_, n0 + j * 16, k0), acc[j]);
  }

  int lane = lane_id();
  int c16 = lane & 15, hi = lane >> 4;
  float bl[8];
#pragma unroll
  for (int r = 0; r < 8; ++r) bl[r] = bias[m0 + hi * 8 + r];

  if (mat < 2) {
    _Float16* dst = (mat == 0 ? qt : kt) + (size_t)b * L_ * CP_;
    float sc = (mat == 0) ? 0.0625f : 1.0f;  // fold 1/sqrt(CP) into Q
#pragma unroll
    for (int j = 0; j < 4; ++j) {
      int l = n0 + j * 16 + c16;
      union { _Float16 h[8]; uint4 u; } pk;
#pragma unroll
      for (int r = 0; r < 8; ++r)
        pk.h[r] = (_Float16)((acc[j][r] + bl[r]) * sc);
      *(uint4*)(dst + (size_t)l * CP_ + m0 + hi * 8) = pk.u;
    }
  } else {
    _Float16* dst = vv + (size_t)b * CP_ * L_;
#pragma unroll
    for (int j = 0; j < 4; ++j) {
      int l = n0 + j * 16 + c16;
#pragma unroll
      for (int r = 0; r < 8; ++r)
        dst[(size_t)(m0 + hi * 8 + r) * L_ + l] = (_Float16)(acc[j][r] + bl[r]);
    }
  }
}

// ---------------------------------------------------------------------------
// Kernel 4: flash attention.
//   S[l,m]   = Qs^T K + log(pm[m]+1e-6)       (Qs already scaled)
//   P        = exp(S - rowmax), denom = sum P  (pre-mask, matches reference)
//   O[l,:]  += (P * pm[m]) @ V^T ; O /= denom ; relu ; store O^T (L,CP) f16
//   4 waves/block, each wave owns 16 rows of L and a 16x256 f32 accumulator.
// ---------------------------------------------------------------------------
__global__ __launch_bounds__(128) void flash_attn_kernel(
    const _Float16* __restrict__ qt, const _Float16* __restrict__ kt,
    const _Float16* __restrict__ vv, const float* __restrict__ mask,
    _Float16* __restrict__ ot) {
  __shared__ _Float16 psc[4][16 * 32];  // wave-private P staging (D->A relayout)

  int w = (int)(threadIdx.x >> 5);
  int b = (int)(blockIdx.x >> 5);
  int l0 = ((int)blockIdx.x & 31) * 64 + w * 16;

  const _Float16* Q = qt + (size_t)b * L_ * CP_;
  const _Float16* K = kt + (size_t)b * L_ * CP_;
  const _Float16* V = vv + (size_t)b * CP_ * L_;
  const float* pm = mask + (size_t)b * L_;
  _Float16* pS = psc[w];

  int lane = lane_id();
  int c16 = lane & 15, hi = lane >> 4;

  float8 accO[16] = {};
  float run_max[8], run_sum[8];
#pragma unroll
  for (int r = 0; r < 8; ++r) { run_max[r] = -__builtin_inff(); run_sum[r] = 0.f; }

  for (int m0 = 0; m0 < L_; m0 += 32) {
    if (m0 + 32 < L_)  // stream hint for next K tile -> global_prefetch_b8
      __builtin_prefetch(K + (size_t)(m0 + 32 + c16) * CP_, 0, 1);

    float pm0 = pm[m0 + c16];
    float pm1 = pm[m0 + 16 + c16];
    float lp0 = __logf(pm0 + 1e-6f);
    float lp1 = __logf(pm1 + 1e-6f);

    float8 s0 = {}, s1 = {};
#pragma unroll
    for (int c0 = 0; c0 < CP_; c0 += 32) {
      half16 a = load_a_frag(Q, CP_, l0, c0);
      s0 = wmma_f16(a, load_b_frag(K, CP_, m0,      c0), s0);
      s1 = wmma_f16(a, load_b_frag(K, CP_, m0 + 16, c0), s1);
    }

    float nm[8];
#pragma unroll
    for (int r = 0; r < 8; ++r) {
      s0[r] += lp0; s1[r] += lp1;
      float t = fmaxf(s0[r], s1[r]);
      t = fmaxf(t, __shfl_xor(t, 1));
      t = fmaxf(t, __shfl_xor(t, 2));
      t = fmaxf(t, __shfl_xor(t, 4));
      t = fmaxf(t, __shfl_xor(t, 8));
      nm[r] = fmaxf(run_max[r], t);
    }
#pragma unroll
    for (int r = 0; r < 8; ++r) {
      float alpha = __expf(run_max[r] - nm[r]);
      run_max[r] = nm[r];
      run_sum[r] *= alpha;
#pragma unroll
      for (int t = 0; t < 16; ++t) accO[t][r] *= alpha;
    }
#pragma unroll
    for (int r = 0; r < 8; ++r) {
      float p0 = __expf(s0[r] - nm[r]);
      float p1 = __expf(s1[r] - nm[r]);
      float rs = p0 + p1;
      rs += __shfl_xor(rs, 1);
      rs += __shfl_xor(rs, 2);
      rs += __shfl_xor(rs, 4);
      rs += __shfl_xor(rs, 8);
      run_sum[r] += rs;
      int lrow = r + 8 * hi;                       // D-layout row for this lane
      pS[lrow * 32 + c16]      = (_Float16)(p0 * pm0);
      pS[lrow * 32 + 16 + c16] = (_Float16)(p1 * pm1);
    }
    // re-read P in A-operand layout (compiler inserts the dscnt waits)
    half16 pa = load_a_frag(pS, 32, 0, 0);
#pragma unroll
    for (int t = 0; t < 16; ++t)
      accO[t] = wmma_f16(pa, load_b_frag(V, L_, t * 16, m0), accO[t]);
  }

  float inv[8];
#pragma unroll
  for (int r = 0; r < 8; ++r) inv[r] = 1.0f / run_sum[r];
  _Float16* O = ot + (size_t)b * L_ * CP_;
#pragma unroll
  for (int t = 0; t < 16; ++t) {
#pragma unroll
    for (int r = 0; r < 8; ++r) {
      float o = fmaxf(accO[t][r] * inv[r], 0.0f);  // relu fused here
      O[(size_t)(l0 + r + 8 * hi) * CP_ + t * 16 + c16] = (_Float16)o;
    }
  }
}

// ---------------------------------------------------------------------------
// Kernel 5: output projection.
//   out[c,l] = (sum_cp wo[c,cp] * Ot[l,cp] + bo[c]) * pm[l]   (f32 out)
// ---------------------------------------------------------------------------
__global__ __launch_bounds__(256) void out_gemm_kernel(
    const _Float16* __restrict__ ot, const _Float16* __restrict__ woh,
    const float* __restrict__ bo, const float* __restrict__ mask,
    float* __restrict__ out) {
  int w = (int)((blockIdx.x * blockDim.x + threadIdx.x) >> 5);
  int nt = w & 31;  w >>= 5;   // L/64  = 32
  int mt = w & 31;  w >>= 5;   // C/16  = 32
  int b  = w;                  // batch

  const _Float16* O = ot + (size_t)b * L_ * CP_;
  int m0 = mt * 16, n0 = nt * 64;

  float8 acc[4] = {};
  for (int k0 = 0; k0 < CP_; k0 += 32) {
    half16 a = load_a_frag(woh, CP_, m0, k0);
#pragma unroll
    for (int j = 0; j < 4; ++j)
      acc[j] = wmma_f16(a, load_b_frag(O, CP_, n0 + j * 16, k0), acc[j]);
  }

  int lane = lane_id();
  int c16 = lane & 15, hi = lane >> 4;
#pragma unroll
  for (int j = 0; j < 4; ++j) {
    int l = n0 + j * 16 + c16;
    float pmv = mask[(size_t)b * L_ + l];
#pragma unroll
    for (int r = 0; r < 8; ++r) {
      int m = m0 + hi * 8 + r;
      out[((size_t)b * C_ + m) * L_ + l] = (acc[j][r] + bo[m]) * pmv;
    }
  }
}

// ---------------------------------------------------------------------------
// Host launcher
// ---------------------------------------------------------------------------
extern "C" void kernel_launch(void* const* d_in, const int* in_sizes, int n_in,
                              void* d_out, int out_size, void* d_ws,
                              size_t ws_size, hipStream_t stream) {
  (void)in_sizes; (void)n_in; (void)out_size; (void)ws_size;

  const float* x1   = (const float*)d_in[0];
  // d_in[1] = x2 (unused by reference)
  const float* mask = (const float*)d_in[2];
  const float* wq   = (const float*)d_in[3];
  const float* bq   = (const float*)d_in[4];
  const float* wk   = (const float*)d_in[5];
  const float* bk   = (const float*)d_in[6];
  const float* wv   = (const float*)d_in[7];
  const float* bv   = (const float*)d_in[8];
  const float* wo   = (const float*)d_in[9];
  const float* bo   = (const float*)d_in[10];
  float* out = (float*)d_out;

  // workspace layout (f16 elements), total ~49 MB
  _Float16* ws = (_Float16*)d_ws;
  const size_t nw = (size_t)CP_ * C_;          // 131072 (== C*CP)
  size_t off = 0;
  _Float16* wh = ws + off; off += 4 * nw;      // [wq|wk|wv|wo] f16
  _Float16* xt = ws + off; off += (size_t)B_ * L_ * C_;   // x^T  (B,L,C)
  _Float16* qt = ws + off; off += (size_t)B_ * L_ * CP_;  // Q^T  (B,L,CP), scaled
  _Float16* kt = ws + off; off += (size_t)B_ * L_ * CP_;  // K^T  (B,L,CP)
  _Float16* vv = ws + off; off += (size_t)B_ * CP_ * L_;  // V    (B,CP,L)
  _Float16* ot = ws + off; off += (size_t)B_ * L_ * CP_;  // O^T  (B,L,CP)

  const int cvt_blk = 256;
  const int cvt_grd = (int)((nw + cvt_blk - 1) / cvt_blk);
  cvt_f32_f16_kernel<<<cvt_grd, cvt_blk, 0, stream>>>(wq, wh + 0 * nw, (int)nw);
  cvt_f32_f16_kernel<<<cvt_grd, cvt_blk, 0, stream>>>(wk, wh + 1 * nw, (int)nw);
  cvt_f32_f16_kernel<<<cvt_grd, cvt_blk, 0, stream>>>(wv, wh + 2 * nw, (int)nw);
  cvt_f32_f16_kernel<<<cvt_grd, cvt_blk, 0, stream>>>(wo, wh + 3 * nw, (int)nw);

  dim3 tb(32, 8);
  dim3 tg(L_ / 32, C_ / 32, B_);
  transpose_x_kernel<<<tg, tb, 0, stream>>>(x1, xt);

  // 3 mats * 8 batch * 16 * 32 strips = 12288 waves / 8 per block
  qkv_gemm_kernel<<<12288 / 8, 256, 0, stream>>>(xt, wh, bq, bk, bv, qt, kt, vv);

  // 8 batch * 32 l-blocks, 4 waves (64 rows) per block
  flash_attn_kernel<<<B_ * (L_ / 64), 128, 0, stream>>>(qt, kt, vv, mask, ot);

  // 8 * 32 * 32 strips = 8192 waves / 8 per block
  out_gemm_kernel<<<8192 / 8, 256, 0, stream>>>(ot, wh + 3 * nw, bo, mask, out);
}
